// bilateral_upsample_edgeConv_22797686407580
// MI455X (gfx1250) — matrix operand order repaired and checked by reading, not compiled, not run on hardware
//
#include <hip/hip_runtime.h>

// ---------------------------------------------------------------------------
// Types for CDNA5 WMMA (wave32, v_wmma_f32_16x16x32_bf16)
// ---------------------------------------------------------------------------
typedef __attribute__((ext_vector_type(16))) __bf16 v16bf;
typedef __attribute__((ext_vector_type(8)))  float  v8f;

union FragBF { unsigned u[8]; v16bf v; };

#define DINF 3.0e38f

__device__ __forceinline__ float lrelu(float x) { return x >= 0.f ? x : 0.01f * x; }

__device__ __forceinline__ unsigned pack2bf(float a, float b) {
    union { __bf16 h[2]; unsigned u; } t;
    t.h[0] = (__bf16)a; t.h[1] = (__bf16)b;
    return t.u;
}

__device__ __forceinline__ void bn_coef(const float* st, const float* g, const float* be,
                                        int Cc, int c, float cnt, float& sc, float& sh) {
    float mean = st[c] / cnt;
    float var  = st[Cc + c] / cnt - mean * mean;
    sc = g[c] * rsqrtf(var + 1e-5f);
    sh = be[c] - mean * sc;
}

// ---------------------------------------------------------------------------
// Row squared norms: norms[b*N+n] = sum_c x[b,c,n]^2
// ---------------------------------------------------------------------------
__global__ void norms_kernel(const float* __restrict__ x, float* __restrict__ norms) {
    int gid = blockIdx.x * blockDim.x + threadIdx.x;
    if (gid >= 8 * 2048) return;
    int b = gid >> 11, n = gid & 2047;
    const float* xb = x + (size_t)b * 64 * 2048 + n;
    float s = 0.f;
    #pragma unroll 8
    for (int c = 0; c < 64; ++c) { float v = xb[(size_t)c * 2048]; s += v * v; }
    norms[gid] = s;
}

// ---------------------------------------------------------------------------
// kNN: one wave per 16 query rows; distance tiles via WMMA (bf16, K=64),
// per-row top-16 kept sorted in registers.
// ---------------------------------------------------------------------------
__global__ void knn_kernel(const float* __restrict__ x, const float* __restrict__ norms,
                           int* __restrict__ idx) {
    __shared__ float dt[16][17];
    int blk = blockIdx.x;
    int b  = blk >> 7;          // N/16 = 128 row tiles per batch
    int rt = blk & 127;
    int n0 = rt * 16;
    int lane  = threadIdx.x;
    int col   = lane & 15;
    int rbase = (lane >> 4) * 8;
    int nrow  = n0 + col;       // A-fragment row owned by this lane
    const float* xb = x + (size_t)b * 64 * 2048;

    // Preload A fragments (16x64 query block -> 2 K-steps of 32)
    int koffA = (lane < 16) ? 0 : 8;
    FragBF fa[2];
    #pragma unroll
    for (int kt = 0; kt < 2; ++kt) {
        int kb = kt * 32 + koffA;
        #pragma unroll
        for (int j = 0; j < 4; ++j) {
            fa[kt].u[j]     = pack2bf(xb[(size_t)(kb + 2*j)     * 2048 + nrow],
                                      xb[(size_t)(kb + 2*j + 1) * 2048 + nrow]);
            fa[kt].u[4 + j] = pack2bf(xb[(size_t)(kb + 16 + 2*j)     * 2048 + nrow],
                                      xb[(size_t)(kb + 16 + 2*j + 1) * 2048 + nrow]);
        }
    }
    const float* nb = norms + (size_t)b * 2048;
    float rn[8];
    #pragma unroll
    for (int r = 0; r < 8; ++r) rn[r] = nb[n0 + rbase + r];

    float bestd[16]; int besti[16];
    #pragma unroll
    for (int j = 0; j < 16; ++j) { bestd[j] = DINF; besti[j] = 0; }

    int koffB = (lane < 16) ? 0 : 8;
    for (int mt = 0; mt < 128; ++mt) {
        int m0 = mt * 16;
        int m  = m0 + col;
        v8f acc = {};
        #pragma unroll
        for (int kt = 0; kt < 2; ++kt) {
            FragBF fb;
            #pragma unroll
            for (int v = 0; v < 8; ++v) {
                int K0 = kt * 32 + koffB + ((v < 4) ? 2 * v : 16 + 2 * (v - 4));
                fb.u[v] = pack2bf(xb[(size_t)K0 * 2048 + m], xb[(size_t)(K0 + 1) * 2048 + m]);
            }
            acc = __builtin_amdgcn_wmma_f32_16x16x32_bf16(false, fa[kt].v, false, fb.v,
                                                          (short)0, acc, false, false);
        }
        float nm = nb[m];
        #pragma unroll
        for (int r = 0; r < 8; ++r) {
            float d = -2.f * acc[r] + rn[r] + nm;
            if (m == n0 + rbase + r) d = DINF;   // exclude self
            dt[rbase + r][col] = d;
        }
        __syncthreads();
        if (lane < 16) {
            #pragma unroll
            for (int j = 0; j < 16; ++j) {
                float d = dt[lane][j];
                if (d < bestd[15]) {
                    bestd[15] = d; besti[15] = m0 + j;
                    #pragma unroll
                    for (int t = 15; t > 0; --t) {
                        if (bestd[t] < bestd[t - 1]) {
                            float td = bestd[t]; bestd[t] = bestd[t-1]; bestd[t-1] = td;
                            int   ti = besti[t]; besti[t] = besti[t-1]; besti[t-1] = ti;
                        }
                    }
                }
            }
        }
        __syncthreads();
    }
    if (lane < 16) {
        int* op = idx + ((size_t)b * 2048 + n0 + lane) * 16;
        #pragma unroll
        for (int j = 0; j < 16; ++j) op[j] = besti[j];
    }
}

// ---------------------------------------------------------------------------
// Edge features: e_fea[b,n,k,c] (bf16, c-inner so im2col windows are contiguous)
// ---------------------------------------------------------------------------
__global__ void edge_fea_kernel(const float* __restrict__ x, const int* __restrict__ idx,
                                __bf16* __restrict__ efea) {
    int gid = blockIdx.x * blockDim.x + threadIdx.x;   // P*128 exactly
    int c  = gid & 127;
    int p  = gid >> 7;
    int pn = p >> 4;
    int b = pn >> 11, n = pn & 2047;
    const float* xb = x + (size_t)b * 64 * 2048;
    float val;
    if (c < 64) {
        val = xb[(size_t)c * 2048 + n];
    } else {
        int m = idx[p]; int cc = c - 64;
        val = xb[(size_t)cc * 2048 + m] - xb[(size_t)cc * 2048 + n];
    }
    efea[gid] = (__bf16)val;
}

__global__ void edge_xyz_kernel(const float* __restrict__ pc, const int* __restrict__ idx,
                                float* __restrict__ exyz) {
    int gid = blockIdx.x * blockDim.x + threadIdx.x;   // P*6 exactly
    int c = gid % 6; int p = gid / 6;
    int pn = p >> 4; int b = pn >> 11, n = pn & 2047;
    const float* pb = pc + (size_t)b * 3 * 2048;
    float val;
    if (c < 3) {
        val = pb[(size_t)c * 2048 + n];
    } else {
        int m = idx[p]; int cc = c - 3;
        val = pb[(size_t)cc * 2048 + m] - pb[(size_t)cc * 2048 + n];
    }
    exyz[gid] = val;
}

// ---------------------------------------------------------------------------
// Weight pack: pre-arrange W[K,O] into exact WMMA B-fragment order.
// dst u32 entry (nt-major, kt, then v*32+lane): pair (K0, K0+1) for col o.
// mode 0: src[o*Ksrc + K] (zero-padded K>=Ksrc)
// mode 1: inte  W[K=kw*128+c, o] = src[(o*128+c)*9  + kw]
// mode 2: c2    W[K=w*128+c,  o] = src[(o*128+c)*32 + w]
// ---------------------------------------------------------------------------
__device__ __forceinline__ float fetchW(const float* src, int K, int o, int Ksrc, int mode) {
    if (mode == 0) return (K < Ksrc) ? src[(size_t)o * Ksrc + K] : 0.f;
    if (mode == 1) { int c = K & 127; int kw = K >> 7; return src[((size_t)o * 128 + c) * 9  + kw]; }
    { int c = K & 127; int w = K >> 7; return src[((size_t)o * 128 + c) * 32 + w]; }
}

__global__ void pack_kernel(const float* __restrict__ src, unsigned* __restrict__ dst,
                            int Ktot, int O, int Ksrc, int mode) {
    int e = blockIdx.x * blockDim.x + threadIdx.x;
    int Kt = Ktot >> 5;
    int total = Kt * (O >> 4) * 256;
    if (e >= total) return;
    int nt = e / (Kt * 256);
    int r  = e - nt * (Kt * 256);
    int kt = r >> 8;
    int q  = r & 255;
    int v  = q >> 5;
    int L  = q & 31;
    int o  = nt * 16 + (L & 15);
    int K0 = kt * 32 + ((L < 16) ? 0 : 8) + ((v < 4) ? 2 * v : 16 + 2 * (v - 4));
    dst[e] = pack2bf(fetchW(src, K0, o, Ksrc, mode), fetchW(src, K0 + 1, o, Ksrc, mode));
}

// ---------------------------------------------------------------------------
// Generic WMMA GEMM, N-blocked: each wave computes NT 16x16 output tiles,
// sharing one A fragment per K-step across NT back-to-back v_wmma issues.
// mode 0: A0 row-major [P, Ktot]
// mode 1: inte im2col, row base = (p>>3)*2048 + (p&7)*128 (overlapping windows)
// mode 2: c2 split K: first 2048 from A0[p*2048..], rest from A1[p*2048..]
// ---------------------------------------------------------------------------
template <int NT>
__global__ void gemm_wmma_t(const __bf16* __restrict__ A0, const __bf16* __restrict__ A1,
                            const unsigned* __restrict__ Wpk, const float* __restrict__ bias,
                            float* __restrict__ Cout, int P, int Ktot, int O, int mode) {
    int numNG = (O >> 4) / NT;
    int tile  = blockIdx.x;
    int pt = tile / numNG, ng = tile - pt * numNG;
    int ntBase = ng * NT;
    int lane = threadIdx.x & 31;
    int p = pt * 16 + (lane & 15);
    size_t base;
    if (mode == 1)      base = (size_t)(p >> 3) * 2048 + (size_t)(p & 7) * 128;
    else if (mode == 2) base = (size_t)p * 2048;
    else                base = (size_t)p * (size_t)Ktot;
    int Kt   = Ktot >> 5;
    int half = Kt >> 1;
    int koffA = (lane < 16) ? 0 : 8;
    const unsigned* wq = Wpk + (size_t)ntBase * Kt * 256 + lane;
    v8f acc[NT] = {};
    for (int kt = 0; kt < Kt; ++kt) {
        const __bf16* Arow; int kofs;
        if (mode == 2 && kt >= half) { Arow = A1 + base; kofs = (kt - half) * 32; }
        else                         { Arow = A0 + base; kofs = kt * 32; }
        const __bf16* pA = Arow + kofs + koffA;
        FragBF fa;
        #pragma unroll
        for (int j = 0; j < 4; ++j) {
            fa.u[j]     = *reinterpret_cast<const unsigned*>(pA + 2 * j);
            fa.u[4 + j] = *reinterpret_cast<const unsigned*>(pA + 16 + 2 * j);
        }
        if (kt + 1 < Kt) {
            // gfx1250 speculative prefetch of next K-step of this A row
            __builtin_prefetch((const void*)(pA + 32), 0, 2);
        }
        #pragma unroll
        for (int i = 0; i < NT; ++i) {
            const unsigned* wk = wq + ((size_t)i * Kt + kt) * 256;
            FragBF fb;
            #pragma unroll
            for (int v = 0; v < 8; ++v) fb.u[v] = wk[v * 32];
            acc[i] = __builtin_amdgcn_wmma_f32_16x16x32_bf16(false, fa.v, false, fb.v,
                                                             (short)0, acc[i], false, false);
        }
    }
    int col   = lane & 15;
    int rbase = (lane >> 4) * 8;
    #pragma unroll
    for (int i = 0; i < NT; ++i) {
        float bv = bias[(ntBase + i) * 16 + col];
        #pragma unroll
        for (int r = 0; r < 8; ++r) {
            Cout[(size_t)(pt * 16 + rbase + r) * O + (ntBase + i) * 16 + col] = acc[i][r] + bv;
        }
    }
}

// ---------------------------------------------------------------------------
// xyz 1x1 conv (K=6, scalar is fine)
// ---------------------------------------------------------------------------
__global__ void conv_xyz_kernel(const float* __restrict__ exyz, const float* __restrict__ w,
                                const float* __restrict__ bias, float* __restrict__ out) {
    int gid = blockIdx.x * blockDim.x + threadIdx.x;   // P*16
    int o = gid & 15; int p = gid >> 4;
    const float* e  = exyz + (size_t)p * 6;
    const float* wr = w + o * 6;
    float s = bias[o];
    #pragma unroll
    for (int c = 0; c < 6; ++c) s += e[c] * wr[c];
    out[gid] = s;
}

// ---------------------------------------------------------------------------
// BN statistics: deterministic block-per-channel LDS tree reduction
// st[c]=sum, st[C+c]=sumsq
// ---------------------------------------------------------------------------
__global__ void stats_kernel(const float* __restrict__ X, int P, int Cc, float* __restrict__ st) {
    __shared__ float s1[256], s2[256];
    int c = blockIdx.x;
    float a = 0.f, b = 0.f;
    for (int p = threadIdx.x; p < P; p += 256) {
        float v = X[(size_t)p * Cc + c];
        a += v; b += v * v;
    }
    s1[threadIdx.x] = a; s2[threadIdx.x] = b;
    __syncthreads();
    for (int s = 128; s > 0; s >>= 1) {
        if (threadIdx.x < s) { s1[threadIdx.x] += s1[threadIdx.x + s]; s2[threadIdx.x] += s2[threadIdx.x + s]; }
        __syncthreads();
    }
    if (threadIdx.x == 0) { st[c] = s1[0]; st[Cc + c] = s2[0]; }
}

// lrelu(bn(fea)) * lrelu(bn(xyz)) -> bf16 [P,32] (upper 16 zero-padded for K=32 WMMA)
__global__ void bnmul_kernel(const float* __restrict__ cf, const float* __restrict__ cx,
                             const float* __restrict__ stF, const float* __restrict__ gF, const float* __restrict__ beF,
                             const float* __restrict__ stX, const float* __restrict__ gX, const float* __restrict__ beX,
                             float cnt, __bf16* __restrict__ out) {
    int gid = blockIdx.x * blockDim.x + threadIdx.x;   // P*32
    int c = gid & 31; int p = gid >> 5;
    float v = 0.f;
    if (c < 16) {
        float sc, sh;
        bn_coef(stF, gF, beF, 16, c, cnt, sc, sh);
        float a = lrelu(cf[(size_t)p * 16 + c] * sc + sh);
        bn_coef(stX, gX, beX, 16, c, cnt, sc, sh);
        float xx = lrelu(cx[(size_t)p * 16 + c] * sc + sh);
        v = a * xx;
    }
    out[gid] = (__bf16)v;
}

// generic bn + lrelu -> bf16 (channel-inner layout)
__global__ void bnapply_kernel(const float* __restrict__ X, const float* __restrict__ st,
                               const float* __restrict__ g, const float* __restrict__ be,
                               int Cc, float cnt, __bf16* __restrict__ out) {
    int gid = blockIdx.x * blockDim.x + threadIdx.x;
    int c = gid % Cc;
    float sc, sh;
    bn_coef(st, g, be, Cc, c, cnt, sc, sh);
    out[gid] = (__bf16)lrelu(X[gid] * sc + sh);
}

// bn + lrelu + softmax over k (stride-128 axis) -> bf16 weights
__global__ void softmax_kernel(const float* __restrict__ X, const float* __restrict__ st,
                               const float* __restrict__ g, const float* __restrict__ be,
                               float cnt, __bf16* __restrict__ out) {
    int gid = blockIdx.x * blockDim.x + threadIdx.x;   // PN*128
    int c = gid & 127; int pn = gid >> 7;
    float sc, sh;
    bn_coef(st, g, be, 128, c, cnt, sc, sh);
    size_t base = (size_t)pn * 16 * 128 + c;
    float v[16]; float mx = -DINF;
    #pragma unroll
    for (int kk = 0; kk < 16; ++kk) {
        v[kk] = lrelu(X[base + (size_t)kk * 128] * sc + sh);
        mx = fmaxf(mx, v[kk]);
    }
    float s = 0.f;
    #pragma unroll
    for (int kk = 0; kk < 16; ++kk) { v[kk] = expf(v[kk] - mx); s += v[kk]; }
    float inv = 1.f / s;
    #pragma unroll
    for (int kk = 0; kk < 16; ++kk) out[base + (size_t)kk * 128] = (__bf16)(v[kk] * inv);
}

// torch reshape dance + bn + lrelu + multiply by softmax weights -> inte_x bf16 [b,n,k,c]
__global__ void intex_kernel(const float* __restrict__ conv_inte, const __bf16* __restrict__ wsoft,
                             const float* __restrict__ st, const float* __restrict__ g,
                             const float* __restrict__ be, float cnt, __bf16* __restrict__ out) {
    int gid = blockIdx.x * blockDim.x + threadIdx.x;   // P*128
    int c = gid & 127; int p = gid >> 7;
    int kk = p & 15; int pn = p >> 4;
    int oc = 2 * c + (kk >> 3);
    float sc, sh;
    bn_coef(st, g, be, 256, oc, cnt, sc, sh);
    float raw = conv_inte[((size_t)pn * 8 + (kk & 7)) * 256 + oc];
    float a = lrelu(raw * sc + sh);
    float w = (float)wsoft[gid];
    out[gid] = (__bf16)(a * w);
}

// final bn + relu + [B,128,N] -> [B,64,2N] reshape
__global__ void final_kernel(const float* __restrict__ conv_c2, const float* __restrict__ st,
                             const float* __restrict__ g, const float* __restrict__ be,
                             float cnt, float* __restrict__ out) {
    int gid = blockIdx.x * blockDim.x + threadIdx.x;   // PN*128
    int c = gid & 127; int pn = gid >> 7;
    int b = pn >> 11, n = pn & 2047;
    float sc, sh;
    bn_coef(st, g, be, 128, c, cnt, sc, sh);
    float v = fmaxf(conv_c2[gid] * sc + sh, 0.f);
    out[((size_t)(b * 64 + (c >> 1))) * 4096 + (size_t)(c & 1) * 2048 + n] = v;
}

// ---------------------------------------------------------------------------
// Host launcher
// ---------------------------------------------------------------------------
extern "C" void kernel_launch(void* const* d_in, const int* in_sizes, int n_in,
                              void* d_out, int out_size, void* d_ws, size_t ws_size,
                              hipStream_t stream) {
    (void)in_sizes; (void)n_in; (void)out_size; (void)ws_size;
    const float* x      = (const float*)d_in[0];
    const float* pc     = (const float*)d_in[1];
    const float* fea_w  = (const float*)d_in[2];
    const float* fea_b  = (const float*)d_in[3];
    const float* fea_g  = (const float*)d_in[4];
    const float* fea_be = (const float*)d_in[5];
    const float* xyz_w  = (const float*)d_in[6];
    const float* xyz_b  = (const float*)d_in[7];
    const float* xyz_g  = (const float*)d_in[8];
    const float* xyz_be = (const float*)d_in[9];
    const float* all1_w = (const float*)d_in[10];
    const float* all1_b = (const float*)d_in[11];
    const float* all1_g = (const float*)d_in[12];
    const float* all1_be= (const float*)d_in[13];
    const float* all2_w = (const float*)d_in[14];
    const float* all2_b = (const float*)d_in[15];
    const float* all2_g = (const float*)d_in[16];
    const float* all2_be= (const float*)d_in[17];
    const float* inte_w = (const float*)d_in[18];
    const float* inte_b = (const float*)d_in[19];
    const float* inte_g = (const float*)d_in[20];
    const float* inte_be= (const float*)d_in[21];
    const float* c2_w   = (const float*)d_in[22];
    const float* c2_b   = (const float*)d_in[23];
    const float* c2_g   = (const float*)d_in[24];
    const float* c2_be  = (const float*)d_in[25];

    const int P  = 8 * 2048 * 16;   // 262144 edge positions
    const int P8 = 8 * 2048 * 8;    // 131072 inte positions
    const int PN = 8 * 2048;        // 16384

    char* ws = (char*)d_ws;
    size_t off = 0;
    auto take = [&](size_t bytes) -> void* {
        off = (off + 255) & ~(size_t)255;
        void* r = ws + off;
        off += bytes;
        return r;
    };

    float*    norms     = (float*)   take((size_t)PN * 4);
    int*      idx       = (int*)     take((size_t)P * 4);
    __bf16*   e_fea     = (__bf16*)  take((size_t)P * 128 * 2);
    float*    e_xyz     = (float*)   take((size_t)P * 6 * 4);
    float*    conv_fea  = (float*)   take((size_t)P * 16 * 4);
    float*    conv_xyz  = (float*)   take((size_t)P * 16 * 4);
    __bf16*   wprod     = (__bf16*)  take((size_t)P * 32 * 2);
    float*    conv_all1 = (float*)   take((size_t)P * 64 * 4);
    __bf16*   w64       = (__bf16*)  take((size_t)P * 64 * 2);
    float*    conv_all2 = (float*)   take((size_t)P * 128 * 4);   // aliased by conv_inte later
    __bf16*   wsoft     = (__bf16*)  take((size_t)P * 128 * 2);
    __bf16*   inte_x    = (__bf16*)  take((size_t)P * 128 * 2);
    float*    conv_c2   = (float*)   take((size_t)PN * 128 * 4);
    unsigned* WfeaPk    = (unsigned*)take(1024 * 4);
    unsigned* Wall1Pk   = (unsigned*)take(1024 * 4);
    unsigned* Wall2Pk   = (unsigned*)take(4096 * 4);
    unsigned* WintePk   = (unsigned*)take(147456 * 4);
    unsigned* Wc2Pk     = (unsigned*)take(262144 * 4);
    float*    stF       = (float*)   take(512 * 4);
    float*    stX       = (float*)   take(512 * 4);
    float*    stA1      = (float*)   take(512 * 4);
    float*    stA2      = (float*)   take(512 * 4);
    float*    stI       = (float*)   take(512 * 4);
    float*    stC2      = (float*)   take(512 * 4);
    float*    conv_inte = conv_all2;  // lifetime-disjoint alias (all2 dead after softmax)

    // Pack weights into WMMA B-fragment order
    pack_kernel<<<   4, 256, 0, stream>>>(fea_w,  WfeaPk,  128,  16, 128, 0);
    pack_kernel<<<   4, 256, 0, stream>>>(all1_w, Wall1Pk,  32,  64,  16, 0);
    pack_kernel<<<  16, 256, 0, stream>>>(all2_w, Wall2Pk,  64, 128,  64, 0);
    pack_kernel<<< 576, 256, 0, stream>>>(inte_w, WintePk, 1152, 256,  0, 1);
    pack_kernel<<<1024, 256, 0, stream>>>(c2_w,   Wc2Pk,  4096, 128,  0, 2);

    // kNN graph
    norms_kernel<<<64, 256, 0, stream>>>(x, norms);
    knn_kernel<<<1024, 32, 0, stream>>>(x, norms, idx);

    // edge features
    edge_fea_kernel<<<131072, 256, 0, stream>>>(x, idx, e_fea);
    edge_xyz_kernel<<<  6144, 256, 0, stream>>>(pc, idx, e_xyz);

    // fea conv (WMMA) + xyz conv (scalar), BN stats, combine
    gemm_wmma_t<1><<<16384, 32, 0, stream>>>(e_fea, e_fea, WfeaPk, fea_b, conv_fea, P, 128, 16, 0);
    conv_xyz_kernel<<<16384, 256, 0, stream>>>(e_xyz, xyz_w, xyz_b, conv_xyz);
    stats_kernel<<<16, 256, 0, stream>>>(conv_fea, P, 16, stF);
    stats_kernel<<<16, 256, 0, stream>>>(conv_xyz, P, 16, stX);
    bnmul_kernel<<<32768, 256, 0, stream>>>(conv_fea, conv_xyz, stF, fea_g, fea_be,
                                            stX, xyz_g, xyz_be, (float)P, wprod);

    // all1 (16->64, K padded to 32), 4 column tiles per wave
    gemm_wmma_t<4><<<16384, 32, 0, stream>>>(wprod, wprod, Wall1Pk, all1_b, conv_all1, P, 32, 64, 0);
    stats_kernel<<<64, 256, 0, stream>>>(conv_all1, P, 64, stA1);
    bnapply_kernel<<<65536, 256, 0, stream>>>(conv_all1, stA1, all1_g, all1_be, 64, (float)P, w64);

    // all2 (64->128) + softmax over k
    gemm_wmma_t<4><<<32768, 32, 0, stream>>>(w64, w64, Wall2Pk, all2_b, conv_all2, P, 64, 128, 0);
    stats_kernel<<<128, 256, 0, stream>>>(conv_all2, P, 128, stA2);
    softmax_kernel<<<8192, 256, 0, stream>>>(conv_all2, stA2, all2_g, all2_be, (float)P, wsoft);

    // inte conv: im2col-free WMMA GEMM, K=1152, O=256 (dominant 77 GFLOP)
    gemm_wmma_t<4><<<32768, 32, 0, stream>>>(e_fea, e_fea, WintePk, inte_b, conv_inte, P8, 1152, 256, 1);
    stats_kernel<<<256, 256, 0, stream>>>(conv_inte, P8, 256, stI);
    intex_kernel<<<131072, 256, 0, stream>>>(conv_inte, wsoft, stI, inte_g, inte_be, (float)P8, inte_x);

    // c2: merge contraction, K=4096 split over (e_fea, inte_x)
    gemm_wmma_t<4><<<2048, 32, 0, stream>>>(e_fea, inte_x, Wc2Pk, c2_b, conv_c2, PN, 4096, 128, 2);
    stats_kernel<<<128, 256, 0, stream>>>(conv_c2, PN, 128, stC2);
    final_kernel<<<8192, 256, 0, stream>>>(conv_c2, stC2, c2_g, c2_be, (float)PN, (float*)d_out);
}
